// Ensemble_SRN_45724221833845
// MI455X (gfx1250) — matrix operand order repaired and checked by reading, not compiled, not run on hardware
//
#include <hip/hip_runtime.h>

// CDNA5 / gfx1250 WMMA vector types
typedef __attribute__((ext_vector_type(16))) _Float16 v16h;
typedef __attribute__((ext_vector_type(8)))  _Float16 v8h;
typedef __attribute__((ext_vector_type(8)))  float    v8f;

#define THREADS  256
#define TILE_PTS 128   // 8 waves x 16 rows per workgroup
#define HPAD     72    // 64 + 8 halves pad -> 144B row stride (16B aligned, conflict-free b128)

// gfx1250 async global->LDS copy path (ASYNCcnt). The ROCm builtin for the
// async load has a non-upstream prototype (v4i* params per the compile probe),
// so we emit the instruction with inline asm (GVS form: vdst = per-lane LDS
// byte address, vaddr = 32-bit per-lane offset, saddr = uniform 64-bit base).
// Generic LDS pointers carry the LDS byte address in their low 32 bits.
#if defined(__gfx1250__) && __has_builtin(__builtin_amdgcn_s_wait_asynccnt)
#define USE_ASYNC_LDS 1
#else
#define USE_ASYNC_LDS 0
#endif

// ---------------- routing -------------------------------------------------

__device__ __forceinline__ int route_expert(const float* __restrict__ x, int i,
                                            int g0, int g1, int g2) {
  const float x0 = x[3 * i + 0], x1 = x[3 * i + 1], x2 = x[3 * i + 2];
  int i0 = (int)((x0 + 1.0f) * 0.5f * (float)g0);
  int i1 = (int)((x1 + 1.0f) * 0.5f * (float)g1);
  int i2 = (int)((x2 + 1.0f) * 0.5f * (float)g2);
  i0 = i0 < 0 ? 0 : (i0 > g0 - 1 ? g0 - 1 : i0);
  i1 = i1 < 0 ? 0 : (i1 > g1 - 1 ? g1 - 1 : i1);
  i2 = i2 < 0 ? 0 : (i2 > g2 - 1 ? g2 - 1 : i2);
  return i0 + i1 * g0 + i2 * (g1 * g2);
}

// ws layout (ints): [0..7] counts, [8..15] scatter cursors, [16..23] offsets,
// perm[] starts at byte offset 256.
__global__ void zero_ws_kernel(int* __restrict__ ws) {
  if (threadIdx.x < 24) ws[threadIdx.x] = 0;
}

__global__ void count_kernel(const float* __restrict__ x, const int* __restrict__ grid,
                             int* __restrict__ ws, int N) {
  int i = blockIdx.x * blockDim.x + threadIdx.x;
  if (i >= N) return;
  atomicAdd(&ws[route_expert(x, i, grid[0], grid[1], grid[2])], 1);
}

__global__ void scan_kernel(int* __restrict__ ws) {
  if (blockIdx.x == 0 && threadIdx.x == 0) {
    int acc = 0;
    for (int e = 0; e < 8; ++e) { ws[16 + e] = acc; acc += ws[e]; }
  }
}

__global__ void scatter_kernel(const float* __restrict__ x, const int* __restrict__ grid,
                               int* __restrict__ ws, int* __restrict__ perm, int N) {
  int i = blockIdx.x * blockDim.x + threadIdx.x;
  if (i >= N) return;
  int e    = route_expert(x, i, grid[0], grid[1], grid[2]);
  int slot = atomicAdd(&ws[8 + e], 1);
  perm[ws[16 + e] + slot] = i;
}

// ---------------- expert MLP (WMMA) ---------------------------------------

__global__ __launch_bounds__(THREADS)
void ensemble_mlp_kernel(const float* __restrict__ x,
                         const float* __restrict__ W1, const float* __restrict__ b1,
                         const float* __restrict__ W2, const float* __restrict__ b2,
                         const float* __restrict__ W3, const float* __restrict__ b3,
                         const int* __restrict__ ws, const int* __restrict__ perm,
                         float* __restrict__ y) {
  __shared__ __align__(16) _Float16 sh_h1[TILE_PTS][HPAD];   // layer-1 activations (fp16)
  __shared__ __align__(16) _Float16 sh_w2t[64][HPAD];        // W2^T fp16: [out_col][k]
#if USE_ASYNC_LDS
  __shared__ __align__(16) float sh_w2raw[64 * 64];          // async-staged raw W2 (f32)
#endif
  __shared__ float sh_w1[3][64];
  __shared__ float sh_b1[64], sh_b2[64], sh_w3[64];
  __shared__ float sh_b3;
  __shared__ float sh_out[TILE_PTS];
  __shared__ int   sh_pidx[TILE_PTS];

  const int e          = blockIdx.y;
  const int count      = ws[e];
  const int base       = ws[16 + e];
  const int tile_start = blockIdx.x * TILE_PTS;
  if (tile_start >= count) return;                 // uniform block-wide exit
  int npts = count - tile_start;
  if (npts > TILE_PTS) npts = TILE_PTS;

  const int t = threadIdx.x;

  // ---- stage this expert's weights ----
#if USE_ASYNC_LDS
  {
    // 16KB W2 panel: 256 lanes x 4 async b128 copies (coalesced), overlapped
    // with the layer-1 VALU work below and drained with s_wait_asynccnt.
    const float* gsrc = W2 + (size_t)e * 4096;     // uniform -> SGPR pair
    #pragma unroll
    for (int k = 0; k < 4; ++k) {
      const int elem = (t + k * THREADS) * 4;      // 4 floats = 16B per op
      const uint32_t ldsa = (uint32_t)(uintptr_t)&sh_w2raw[elem];  // LDS byte addr
      const uint32_t voff = (uint32_t)(elem * sizeof(float));      // global byte off
      asm volatile("global_load_async_to_lds_b128 %0, %1, %2"
                   :: "v"(ldsa), "v"(voff), "s"(gsrc)
                   : "memory");
    }
  }
#else
  for (int i = t; i < 64 * 64; i += THREADS)
    sh_w2t[i & 63][i >> 6] = (_Float16)W2[e * 4096 + i];     // W2[e][k][j] -> w2t[j][k]
#endif
  for (int i = t; i < 3 * 64; i += THREADS)
    sh_w1[i >> 6][i & 63] = W1[e * 192 + i];
  if (t < 64) {
    sh_b1[t] = b1[e * 64 + t];
    sh_b2[t] = b2[e * 64 + t];
    sh_w3[t] = W3[e * 64 + t];
  }
  if (t == 0) sh_b3 = b3[e];
  __syncthreads();

  // ---- layer 1 (3 -> 64), scalar VALU, write fp16 rows into LDS ----
  if (t < TILE_PTS) {
    if (t < npts) {
      const int gi = perm[base + tile_start + t];
      sh_pidx[t] = gi;
      const float x0 = x[3 * gi], x1 = x[3 * gi + 1], x2 = x[3 * gi + 2];
      #pragma unroll
      for (int j = 0; j < 64; ++j) {
        float h = fmaf(x0, sh_w1[0][j],
                  fmaf(x1, sh_w1[1][j],
                  fmaf(x2, sh_w1[2][j], sh_b1[j])));
        sh_h1[t][j] = (_Float16)fmaxf(h, 0.0f);
      }
    } else {
      #pragma unroll
      for (int j = 0; j < 64; ++j) sh_h1[t][j] = (_Float16)0.0f;  // pad rows
    }
  }

#if USE_ASYNC_LDS
  __builtin_amdgcn_s_wait_asynccnt(0);             // drain this wave's async copies
  __syncthreads();                                 // cover sh_h1 + sh_w2raw (all waves)
  for (int i = t; i < 64 * 64; i += THREADS)       // LDS f32 -> transposed fp16 panel
    sh_w2t[i & 63][i >> 6] = (_Float16)sh_w2raw[i];
#endif
  __syncthreads();

  // ---- layer 2 (64 -> 64) WMMA, layer 3 (64 -> 1) fused in-fragment ----
  {
    const int wave  = t >> 5;
    const int lane  = t & 31;
    const int lhalf = lane >> 4;          // 0 | 1
    const int l15   = lane & 15;
    const int arow  = wave * 16 + l15;    // A-matrix: row M striped within lane
    const int khalf = lhalf << 3;         // A: lanes 16-31 hold K+8 block

    v16h a0, a1;
    {
      v8h lo0 = *(const v8h*)&sh_h1[arow][khalf];            // K 0..7   (+khalf)
      v8h hi0 = *(const v8h*)&sh_h1[arow][khalf + 16];       // K 16..23 (+khalf)
      v8h lo1 = *(const v8h*)&sh_h1[arow][32 + khalf];       // K 32..39 (+khalf)
      v8h hi1 = *(const v8h*)&sh_h1[arow][32 + khalf + 16];  // K 48..55 (+khalf)
      #pragma unroll
      for (int i = 0; i < 8; ++i) {
        a0[i] = lo0[i]; a0[i + 8] = hi0[i];
        a1[i] = lo1[i]; a1[i + 8] = hi1[i];
      }
    }

    float psum[8];
    #pragma unroll
    for (int r = 0; r < 8; ++r) psum[r] = 0.0f;

    const int kseg = lhalf << 3;          // B: matches A, lanes 16-31 hold K+8 block
    #pragma unroll
    for (int nt = 0; nt < 4; ++nt) {
      const int ncol = nt * 16 + l15;
      v16h bk0, bk1;
      {
        v8h q0 = *(const v8h*)&sh_w2t[ncol][kseg];           // k-step 0
        v8h q1 = *(const v8h*)&sh_w2t[ncol][kseg + 16];
        v8h q2 = *(const v8h*)&sh_w2t[ncol][32 + kseg];      // k-step 1
        v8h q3 = *(const v8h*)&sh_w2t[ncol][32 + kseg + 16];
        #pragma unroll
        for (int i = 0; i < 8; ++i) {
          bk0[i] = q0[i]; bk0[i + 8] = q1[i];
          bk1[i] = q2[i]; bk1[i + 8] = q3[i];
        }
      }
      const float bv = sh_b2[ncol];
      v8f acc;
      #pragma unroll
      for (int i = 0; i < 8; ++i) acc[i] = bv;               // bias preload into C

      acc = __builtin_amdgcn_wmma_f32_16x16x32_f16(
                false, a0, false, bk0, (short)0, acc, false, false);
      acc = __builtin_amdgcn_wmma_f32_16x16x32_f16(
                false, a1, false, bk1, (short)0, acc, false, false);

      // Fused layer 3: relu + multiply by this lane's w3 column, accumulate.
      const float w3c = sh_w3[ncol];
      #pragma unroll
      for (int r = 0; r < 8; ++r)
        psum[r] = fmaf(fmaxf(acc[r], 0.0f), w3c, psum[r]);
    }

    // Butterfly-reduce over the 16-lane N groups (xor bits 0..3 keep halves apart)
    #pragma unroll
    for (int m = 1; m <= 8; m <<= 1) {
      #pragma unroll
      for (int r = 0; r < 8; ++r)
        psum[r] += __shfl_xor(psum[r], m, 32);
    }

    // C layout: VGPR r -> M = r (lanes 0-15) / r+8 (lanes 16-31)
    const int rbase = wave * 16 + (lhalf << 3);
    #pragma unroll
    for (int r = 0; r < 8; ++r)
      if (l15 == r) sh_out[rbase + r] = psum[r] + sh_b3;
  }
  __syncthreads();

  // ---- scatter results ----
  if (t < npts) y[sh_pidx[t]] = sh_out[t];
}

// ---------------- launch ---------------------------------------------------

extern "C" void kernel_launch(void* const* d_in, const int* in_sizes, int n_in,
                              void* d_out, int out_size, void* d_ws, size_t ws_size,
                              hipStream_t stream) {
  const float* x    = (const float*)d_in[0];
  const float* W1   = (const float*)d_in[1];
  const float* b1   = (const float*)d_in[2];
  const float* W2   = (const float*)d_in[3];
  const float* b2   = (const float*)d_in[4];
  const float* W3   = (const float*)d_in[5];
  const float* b3   = (const float*)d_in[6];
  const int*   grid = (const int*)d_in[7];
  float*       y    = (float*)d_out;

  const int N = in_sizes[0] / 3;

  int* wsI  = (int*)d_ws;
  int* perm = (int*)((char*)d_ws + 256);

  const int rblocks = (N + THREADS - 1) / THREADS;

  zero_ws_kernel<<<1, 32, 0, stream>>>(wsI);
  count_kernel<<<rblocks, THREADS, 0, stream>>>(x, grid, wsI, N);
  scan_kernel<<<1, 1, 0, stream>>>(wsI);
  scatter_kernel<<<rblocks, THREADS, 0, stream>>>(x, grid, wsI, perm, N);

  dim3 g((N + TILE_PTS - 1) / TILE_PTS, 8);
  ensemble_mlp_kernel<<<g, THREADS, 0, stream>>>(x, W1, b1, W2, b2, W3, b3,
                                                 wsI, perm, y);
}
